// EdgeDecoder_2310692405394
// MI455X (gfx1250) — matrix operand order repaired and checked by reading, not compiled, not run on hardware
//
#include <hip/hip_runtime.h>
#include <hip/hip_bf16.h>
#include <math.h>

typedef _Float16 v16h __attribute__((ext_vector_type(16)));
typedef _Float16 v8h  __attribute__((ext_vector_type(8)));
typedef float    v8f  __attribute__((ext_vector_type(8)));

#define DIMD     128
#define HID      256
#define MTILE    128
#define XSTRIDE  264     // 256 + 8 halves pad -> conflict-free b128 LDS reads
#define KSTEPS   8       // 256 / 32
#define NTILES   16      // 256 / 16
#define MT       8       // M-tiles per block (128 rows / 16)

// ---------------------------------------------------------------------------
// Prep: pack W1 (f32 row-major [256 x 256], [k][n]) into f16 WMMA B-fragment
// order. Fragment (nt, ks) covers k = ks*32..+31, n = nt*16..+15.
// Consuming-lane layout (B 32x16, 16-bit): n = lane%16, k-half = lane/16,
// element i -> k = ks*32 + (lane/16)*16 + i. Lane data is 16 contiguous f16.
// ---------------------------------------------------------------------------
__global__ void pack_w1_kernel(const float* __restrict__ W1,
                               _Float16* __restrict__ wfrag) {
    int idx  = blockIdx.x * blockDim.x + threadIdx.x;   // 0 .. 65535
    int frag = idx >> 9;          // /512
    int r    = idx & 511;
    int lane = r >> 4;
    int i    = r & 15;
    int nt = frag >> 3;
    int ks = frag & 7;
    int n  = nt * 16 + (lane & 15);
    int k  = ks * 32 + ((lane >> 4) << 4) + i;
    wfrag[idx] = (_Float16)W1[k * HID + n];
}

// ---------------------------------------------------------------------------
// Main: gather+mul -> f16 X tile in LDS -> WMMA GEMM (X @ W1), each wave owns
// 2 N-tiles x all 8 M-tiles (B loaded once per block -> minimal L2 traffic),
// A fragments double-buffered from LDS, fused bias/relu/GEMV(W2)/sigmoid
// epilogue via LDS wave-slice reduction.
// ---------------------------------------------------------------------------
__global__ __launch_bounds__(256) void edge_decoder_kernel(
    const float* __restrict__ z, const float* __restrict__ z2,
    const int* __restrict__ edge, const float* __restrict__ b1,
    const float* __restrict__ W2, const float* __restrict__ b2,
    const _Float16* __restrict__ wfrag, float* __restrict__ out, int E)
{
    __shared__ _Float16 xlds[MTILE * XSTRIDE];   // ~66 KB
    __shared__ float hpart[8 * MTILE];           // per-wave row partials, 4 KB
    __shared__ float b1_lds[HID];
    __shared__ float w2_lds[HID];

    const int tid       = threadIdx.x;
    const int blockBase = blockIdx.x * MTILE;

    // stage b1 / W2 (256 threads, 256 elements each)
    b1_lds[tid] = b1[tid];
    w2_lds[tid] = W2[tid];

    // ---- build X tile: 2 threads per edge row, 128 cols each -------------
    {
        const int row  = tid >> 1;      // 0..127
        const int hsel = tid & 1;       // 0: z-part (cols 0-127), 1: z2-part
        int e = blockBase + row;
        if (e >= E) e = E - 1;          // clamp tail (keeps EXEC full)
        const int s = edge[e];
        const int d = edge[E + e];
        const float* bp = hsel ? z2 : z;
        const float* ps = bp + (size_t)s * DIMD;
        const float* pd = bp + (size_t)d * DIMD;
        _Float16* xr = &xlds[row * XSTRIDE + hsel * DIMD];
        #pragma unroll
        for (int c = 0; c < DIMD; c += 8) {
            float4 a0 = *(const float4*)(ps + c);
            float4 a1 = *(const float4*)(ps + c + 4);
            float4 c0 = *(const float4*)(pd + c);
            float4 c1 = *(const float4*)(pd + c + 4);
            v8h hv;
            hv[0] = (_Float16)(a0.x * c0.x);
            hv[1] = (_Float16)(a0.y * c0.y);
            hv[2] = (_Float16)(a0.z * c0.z);
            hv[3] = (_Float16)(a0.w * c0.w);
            hv[4] = (_Float16)(a1.x * c1.x);
            hv[5] = (_Float16)(a1.y * c1.y);
            hv[6] = (_Float16)(a1.z * c1.z);
            hv[7] = (_Float16)(a1.w * c1.w);
            *(v8h*)(xr + c) = hv;
        }
    }
    __syncthreads();

    const int lane = tid & 31;
    const int wave = tid >> 5;               // 8 waves, each owns N-tiles 2w,2w+1
    const int nlo  = lane & 15;
    const int kb   = lane >> 4;              // A-fragment K-half / C-row-half

    v8f acc[MT][2];
    #pragma unroll
    for (int mt = 0; mt < MT; ++mt) {
        acc[mt][0] = (v8f){};
        acc[mt][1] = (v8f){};
    }

    const _Float16* wbase = wfrag + lane * 16;
    const int nt0 = 2 * wave;
    const _Float16* abase = &xlds[nlo * XSTRIDE + kb * 16];

    // ---- GEMM1: ks-outer, B loaded once per ks, A double-buffered --------
    union AF { v16h v; v8h h[2]; };
    #pragma unroll
    for (int ks = 0; ks < KSTEPS; ++ks) {
        v16h bv0 = *(const v16h*)(wbase + ((nt0    ) * KSTEPS + ks) * 512);
        v16h bv1 = *(const v16h*)(wbase + ((nt0 + 1) * KSTEPS + ks) * 512);
        AF a[2];
        {
            const _Float16* ar = abase + ks * 32;          // mt = 0
            a[0].h[0] = *(const v8h*)(ar);
            a[0].h[1] = *(const v8h*)(ar + 8);
        }
        #pragma unroll
        for (int mt = 0; mt < MT; ++mt) {
            const int cur = mt & 1;
            if (mt + 1 < MT) {                             // prefetch next A
                const _Float16* ar = abase + (mt + 1) * (16 * XSTRIDE) + ks * 32;
                a[cur ^ 1].h[0] = *(const v8h*)(ar);
                a[cur ^ 1].h[1] = *(const v8h*)(ar + 8);
            }
            acc[mt][0] = __builtin_amdgcn_wmma_f32_16x16x32_f16(
                false, a[cur].v, false, bv0, (short)0, acc[mt][0], false, false);
            acc[mt][1] = __builtin_amdgcn_wmma_f32_16x16x32_f16(
                false, a[cur].v, false, bv1, (short)0, acc[mt][1], false, false);
        }
    }

    // ---- fused epilogue: +b1, relu, * W2[n], per-row partials -------------
    float part[MT][8];
    #pragma unroll
    for (int mt = 0; mt < MT; ++mt)
        #pragma unroll
        for (int j = 0; j < 8; ++j) part[mt][j] = 0.0f;

    #pragma unroll
    for (int ntl = 0; ntl < 2; ++ntl) {
        const int n = (nt0 + ntl) * 16 + nlo;
        const float b1v = b1_lds[n];
        const float w2v = w2_lds[n];
        #pragma unroll
        for (int mt = 0; mt < MT; ++mt) {
            #pragma unroll
            for (int j = 0; j < 8; ++j) {
                float h = acc[mt][ntl][j] + b1v;
                h = h > 0.0f ? h : 0.0f;       // relu
                part[mt][j] += h * w2v;        // fold GEMV (H @ W2)
            }
        }
    }

    // 16-lane xor-butterfly allreduce (halves [0,16) and [16,32) independent)
    #pragma unroll
    for (int mt = 0; mt < MT; ++mt) {
        #pragma unroll
        for (int j = 0; j < 8; ++j) {
            float v = part[mt][j];
            v += __shfl_xor(v, 1, 32);
            v += __shfl_xor(v, 2, 32);
            v += __shfl_xor(v, 4, 32);
            v += __shfl_xor(v, 8, 32);
            part[mt][j] = v;
        }
    }

    // C-layout: lanes 0-15 hold rows M=j, lanes 16-31 hold M=8+j.
    // Lanes 0 and 16 write this wave's row sums as float4 (conflict-free).
    if (nlo == 0) {
        float* hp = &hpart[wave * MTILE + (kb << 3)];
        #pragma unroll
        for (int mt = 0; mt < MT; ++mt) {
            float4 lo = make_float4(part[mt][0], part[mt][1], part[mt][2], part[mt][3]);
            float4 hi = make_float4(part[mt][4], part[mt][5], part[mt][6], part[mt][7]);
            *(float4*)(hp + mt * 16)     = lo;
            *(float4*)(hp + mt * 16 + 4) = hi;
        }
    }
    __syncthreads();

    // ---- final: sum 8 wave slices, +b2, sigmoid, store --------------------
    if (tid < MTILE) {
        float s = 0.0f;
        #pragma unroll
        for (int w = 0; w < 8; ++w) s += hpart[w * MTILE + tid];
        const int e = blockBase + tid;
        if (e < E) {
            float x = s + b2[0];
            out[e] = 1.0f / (1.0f + __expf(-x));   // sigmoid
        }
    }
}

// ---------------------------------------------------------------------------
extern "C" void kernel_launch(void* const* d_in, const int* in_sizes, int n_in,
                              void* d_out, int out_size, void* d_ws, size_t ws_size,
                              hipStream_t stream) {
    const float* z    = (const float*)d_in[0];
    const float* z2   = (const float*)d_in[1];
    const int*   edge = (const int*)d_in[2];
    const float* W1   = (const float*)d_in[3];
    const float* b1   = (const float*)d_in[4];
    const float* W2   = (const float*)d_in[5];
    const float* b2   = (const float*)d_in[6];
    float* out = (float*)d_out;

    const int E = in_sizes[2] / 2;               // edge is [2, E]
    _Float16* wfrag = (_Float16*)d_ws;           // 256*256 f16 = 128 KB

    // pack W1 into WMMA fragment order (128 fragments * 512 f16 = 65536 elems)
    pack_w1_kernel<<<(NTILES * KSTEPS * 512) / 256, 256, 0, stream>>>(W1, wfrag);

    const int nblocks = (E + MTILE - 1) / MTILE;
    edge_decoder_kernel<<<nblocks, 256, 0, stream>>>(z, z2, edge, b1, W2, b2,
                                                     wfrag, out, E);
}
